// SimpleSSMBlock_65403761983889
// MI455X (gfx1250) — matrix operand order repaired and checked by reading, not compiled, not run on hardware
//
#include <hip/hip_runtime.h>
#include <stdint.h>

// ---------------------------------------------------------------------------
// SimpleSSMBlock for MI455X (gfx1250)
//   - bf16 WMMA GEMMs (v_wmma_f32_16x16x32_bf16), f32 accumulate
//   - async global->LDS DMA staging (GLOBAL_LOAD_ASYNC_TO_LDS_B128, ASYNCcnt)
//   - double-buffered LDS pipeline, one barrier per K-step
//   - fused conv/scan/gate with fast silu (v_rcp_f32 instead of IEEE div)
// ---------------------------------------------------------------------------

typedef __attribute__((ext_vector_type(16))) __bf16 v16bf;
typedef __attribute__((ext_vector_type(8)))  __bf16 v8bf;
typedef __attribute__((ext_vector_type(8)))  float  v8f;

#define DMODEL 1024
#define TLEN   4096
#define BATCH  8
#define NTOK   (BATCH * TLEN)     // 32768 rows
#define LN_EPS 1e-5f

// round-to-nearest-even fp32 -> bf16 (bit pattern)
__device__ __forceinline__ unsigned short f2bf_bits(float f) {
    unsigned int u = __float_as_uint(f);
    u = (u + 0x7FFFu + ((u >> 16) & 1u)) >> 16;
    return (unsigned short)u;
}
__device__ __forceinline__ __bf16 f2bf(float f) {
    unsigned short s = f2bf_bits(f);
    return __builtin_bit_cast(__bf16, s);
}

// fast silu: x * rcp(1+exp(-x)) -> v_exp_f32 + v_rcp_f32, no IEEE div chain
__device__ __forceinline__ float siluf(float x) {
    return x * __builtin_amdgcn_rcpf(1.0f + __expf(-x));
}

// low 32 bits of a flat LDS address == workgroup-relative LDS byte offset
__device__ __forceinline__ uint32_t lds_off(const void* p) {
    return (uint32_t)(uintptr_t)p;
}

// per-lane async DMA: 16B  global[saddr + voff] -> LDS[ldsoff]   (ASYNCcnt)
#define ASYNC_B128(ldsoff, voff, saddr)                                  \
    asm volatile("global_load_async_to_lds_b128 %0, %1, %2"              \
                 :: "v"(ldsoff), "v"(voff), "s"(saddr) : "memory")
#define WAIT_ASYNC0() asm volatile("s_wait_asynccnt 0x0" ::: "memory")

// ---------------------------------------------------------------------------
// x fp32 -> bf16 (packed 4-at-a-time)
// ---------------------------------------------------------------------------
__global__ __launch_bounds__(256) void cvt_x_kernel(const float* __restrict__ x,
                                                    __bf16* __restrict__ xb, int n4) {
    int i = blockIdx.x * blockDim.x + threadIdx.x;
    if (i >= n4) return;
    float4 f = ((const float4*)x)[i];
    unsigned long long pk =
        (unsigned long long)f2bf_bits(f.x)        |
        ((unsigned long long)f2bf_bits(f.y) << 16) |
        ((unsigned long long)f2bf_bits(f.z) << 32) |
        ((unsigned long long)f2bf_bits(f.w) << 48);
    ((unsigned long long*)xb)[i] = pk;
}

// ---------------------------------------------------------------------------
// Weight convert + transpose: W[K][N] fp32 -> Wt[N][K] bf16
// ---------------------------------------------------------------------------
__global__ void cvt_transpose_kernel(const float* __restrict__ W,
                                     __bf16* __restrict__ Wt, int K, int N) {
    int idx = blockIdx.x * blockDim.x + threadIdx.x;
    if (idx >= K * N) return;
    int k = idx / N;
    int n = idx - k * N;                       // coalesced read of W
    Wt[(size_t)n * K + k] = f2bf(W[idx]);
}

// ---------------------------------------------------------------------------
// WMMA GEMM: C[M][N] = A[M][K] * Bt[N][K]^T + bias   (A, Bt both bf16)
//   Block tile 128x128, BK=32, 8 waves (2 x 4), wave tile 64x32 = 4x2 WMMA.
//   Double-buffered LDS; tile k+1 DMA'd via async-to-LDS while tile k runs
//   on the WMMA pipe. One s_wait_asynccnt + one barrier per K-step.
//   EPI 0: N==2048 in-proj epilogue -> u (cols<1024) and silu(v) (cols>=1024)
//   EPI 1: plain fp32 store with bias
// ---------------------------------------------------------------------------
#define BK   32
#define LDK  40   // +8 bf16 pad: 80B row stride keeps 16B align, conflict-free

__device__ __forceinline__ v16bf load_frag_a(const __bf16* s, int row0, int lane) {
    // A 16x32 bf16: lane L -> M=row0+(L%16); elems 0..7: K=(L/16)*8+0..7,
    // elems 8..15: K=16+(L/16)*8+0..7  (two contiguous 16B runs)
    int r  = row0 + (lane & 15);
    int kh = (lane >> 4) * 8;
    const __bf16* p = s + r * LDK;
    union { v16bf v; v8bf h[2]; } t;
    t.h[0] = *(const v8bf*)(p + kh);
    t.h[1] = *(const v8bf*)(p + 16 + kh);
    return t.v;
}

__device__ __forceinline__ v16bf load_frag_b(const __bf16* s, int col0, int lane) {
    // B 32x16 bf16 (stored K-contiguous per column n):
    // lane L -> N=col0+(L%16); elems 0..7: K=(L/16)*8+0..7, elems 8..15: K=16+(L/16)*8+0..7
    int n  = col0 + (lane & 15);
    int k0 = (lane >> 4) * 8;
    const __bf16* p = s + n * LDK + k0;
    union { v16bf v; v8bf h[2]; } t;
    t.h[0] = *(const v8bf*)(p);
    t.h[1] = *(const v8bf*)(p + 16);
    return t.v;
}

template <int EPI>
__global__ __launch_bounds__(256) void gemm_wmma_kernel(
        const __bf16* __restrict__ A, const __bf16* __restrict__ Bt,
        const float* __restrict__ bias,
        float* __restrict__ out0, float* __restrict__ out1,
        int M, int N, int K) {
    __shared__ __bf16 As[2][128 * LDK];
    __shared__ __bf16 Bs[2][128 * LDK];

    const int tid  = threadIdx.x;
    const int lane = tid & 31;
    const int wid  = tid >> 5;
    const int wm   = wid >> 2;        // 0..1  (64 rows each)
    const int wn   = wid & 3;         // 0..3  (32 cols each)
    const int bM   = blockIdx.x * 128;
    const int bN   = blockIdx.y * 128;

    // staging assignment: thread -> rows {r, r+64}, 16B chunk c (8 bf16)
    const int rS = tid >> 2;          // 0..63
    const int cS = (tid & 3) * 8;     // 0,8,16,24

    const uint32_t ldsA0 = lds_off(&As[0][rS * LDK + cS]);
    const uint32_t ldsA1 = lds_off(&As[0][(rS + 64) * LDK + cS]);
    const uint32_t ldsB0 = lds_off(&Bs[0][rS * LDK + cS]);
    const uint32_t ldsB1 = lds_off(&Bs[0][(rS + 64) * LDK + cS]);
    const uint32_t bufStride = (uint32_t)(128 * LDK * sizeof(__bf16));

    // 32-bit global byte offsets (max 64MB buffers -> fits easily)
    uint32_t gA0 = (uint32_t)(((bM + rS) * K + cS) * sizeof(__bf16));
    uint32_t gA1 = (uint32_t)(((bM + rS + 64) * K + cS) * sizeof(__bf16));
    uint32_t gB0 = (uint32_t)(((bN + rS) * K + cS) * sizeof(__bf16));
    uint32_t gB1 = (uint32_t)(((bN + rS + 64) * K + cS) * sizeof(__bf16));
    const uint32_t kStep = (uint32_t)(BK * sizeof(__bf16));

    v8f acc[4][2];
#pragma unroll
    for (int i = 0; i < 4; ++i)
#pragma unroll
        for (int j = 0; j < 2; ++j)
#pragma unroll
            for (int r = 0; r < 8; ++r) acc[i][j][r] = 0.0f;

    // prefetch tile 0 into buffer 0
    ASYNC_B128(ldsA0, gA0, A);
    ASYNC_B128(ldsA1, gA1, A);
    ASYNC_B128(ldsB0, gB0, Bt);
    ASYNC_B128(ldsB1, gB1, Bt);
    gA0 += kStep; gA1 += kStep; gB0 += kStep; gB1 += kStep;

    uint32_t bufOff = 0;
    for (int kt = 0; kt < K; kt += BK) {
        WAIT_ASYNC0();          // my async writes for current tile landed
        __syncthreads();        // everyone's landed; prev-tile reads all done

        if (kt + BK < K) {      // DMA next tile into the other buffer
            uint32_t nb = bufOff ^ bufStride;
            ASYNC_B128(ldsA0 + nb, gA0, A);
            ASYNC_B128(ldsA1 + nb, gA1, A);
            ASYNC_B128(ldsB0 + nb, gB0, Bt);
            ASYNC_B128(ldsB1 + nb, gB1, Bt);
            gA0 += kStep; gA1 += kStep; gB0 += kStep; gB1 += kStep;
        }

        const __bf16* as = (const __bf16*)((const char*)&As[0][0] + bufOff);
        const __bf16* bs = (const __bf16*)((const char*)&Bs[0][0] + bufOff);

        v16bf af[4], bfr[2];
#pragma unroll
        for (int i = 0; i < 4; ++i) af[i] = load_frag_a(as, wm * 64 + i * 16, lane);
#pragma unroll
        for (int j = 0; j < 2; ++j) bfr[j] = load_frag_b(bs, wn * 32 + j * 16, lane);

#pragma unroll
        for (int i = 0; i < 4; ++i)
#pragma unroll
            for (int j = 0; j < 2; ++j)
                acc[i][j] = __builtin_amdgcn_wmma_f32_16x16x32_bf16(
                    false, af[i], false, bfr[j], (short)0, acc[i][j], false, false);

        bufOff ^= bufStride;
    }

    // ---- epilogue: C layout lane L,VGPR r -> M = r + 8*(L/16), N = L%16 ----
    const int mBase = bM + wm * 64;
    const int nBase = bN + wn * 32;
    const int lrow  = (lane >> 4) * 8;
    const int lcol  = lane & 15;
    const bool uHalf = (bN < DMODEL);   // uniform per block (boundary % 128 == 0)
#pragma unroll
    for (int i = 0; i < 4; ++i) {
#pragma unroll
        for (int j = 0; j < 2; ++j) {
            int col = nBase + j * 16 + lcol;
            float bv = bias[col];
#pragma unroll
            for (int r = 0; r < 8; ++r) {
                int row = mBase + i * 16 + lrow + r;
                float val = acc[i][j][r] + bv;
                if (EPI == 0) {
                    if (uHalf)
                        out0[(size_t)row * DMODEL + col] = val;                   // u
                    else
                        out1[(size_t)row * DMODEL + (col - DMODEL)] = siluf(val); // silu(v)
                } else {
                    out0[(size_t)row * N + col] = val;
                }
            }
        }
    }
}

// ---------------------------------------------------------------------------
// Fused depthwise conv (K=3) + SSM scan + silu gate.
// One thread per (batch, channel); loads coalesced across channels.
// z = u * silu(C*h + D*vl), h = a*h + B*vl      -> stored bf16 for GEMM2
// ---------------------------------------------------------------------------
__global__ __launch_bounds__(256) void conv_scan_gate_kernel(
        const float* __restrict__ v, const float* __restrict__ u,
        const float* __restrict__ conv_w, const float* __restrict__ conv_b,
        const float* __restrict__ logit_a, const float* __restrict__ ssm_b,
        const float* __restrict__ ssm_c, const float* __restrict__ ssm_d,
        __bf16* __restrict__ z) {
    int ch = blockIdx.x * blockDim.x + threadIdx.x;    // 0 .. BATCH*DMODEL-1
    if (ch >= BATCH * DMODEL) return;
    int b = ch / DMODEL;
    int d = ch - b * DMODEL;

    const float w0 = conv_w[d * 3 + 0];
    const float w1 = conv_w[d * 3 + 1];
    const float w2 = conv_w[d * 3 + 2];
    const float cb = conv_b[d];
    const float a  = __builtin_amdgcn_rcpf(1.0f + __expf(-logit_a[d]));
    const float Bc = ssm_b[d];
    const float Cc = ssm_c[d];
    const float Dc = ssm_d[d];

    const size_t base = (size_t)b * TLEN * DMODEL + d;
    const float* vp = v + base;
    const float* up = u + base;
    __bf16*      zp = z + base;

    float h   = 0.0f;
    float vm1 = 0.0f;          // v[t-1], zero-padded
    float v0  = vp[0];         // v[t]
#pragma unroll 4
    for (int t = 0; t < TLEN; ++t) {
        float vp1 = (t + 1 < TLEN) ? vp[(size_t)(t + 1) * DMODEL] : 0.0f;
        float vl  = fmaf(w0, vm1, fmaf(w1, v0, fmaf(w2, vp1, cb)));
        h  = fmaf(a, h, Bc * vl);
        float y  = fmaf(Cc, h, Dc * vl);
        float zz = up[(size_t)t * DMODEL] * siluf(y);
        zp[(size_t)t * DMODEL] = f2bf(zz);
        vm1 = v0; v0 = vp1;
    }
}

// ---------------------------------------------------------------------------
// Residual add + LayerNorm over D=1024. One 256-thread block per row.
// ---------------------------------------------------------------------------
__global__ __launch_bounds__(256) void ln_kernel(
        const float* __restrict__ op, const float* __restrict__ x,
        const float* __restrict__ w, const float* __restrict__ b,
        float* __restrict__ out) {
    const int row = blockIdx.x;
    const float* pr = op + (size_t)row * DMODEL;
    const float* px = x  + (size_t)row * DMODEL;
    float hv[4];
    float s = 0.0f, ss = 0.0f;
#pragma unroll
    for (int i = 0; i < 4; ++i) {
        int c = threadIdx.x + i * 256;
        hv[i] = pr[c] + px[c];
        s  += hv[i];
        ss += hv[i] * hv[i];
    }
    __shared__ float rs[256], rss[256];
    rs[threadIdx.x]  = s;
    rss[threadIdx.x] = ss;
    __syncthreads();
    for (int off = 128; off > 0; off >>= 1) {
        if (threadIdx.x < off) {
            rs[threadIdx.x]  += rs[threadIdx.x + off];
            rss[threadIdx.x] += rss[threadIdx.x + off];
        }
        __syncthreads();
    }
    const float mu   = rs[0] * (1.0f / DMODEL);
    const float var  = rss[0] * (1.0f / DMODEL) - mu * mu;
    const float rstd = rsqrtf(var + LN_EPS);
    float* po = out + (size_t)row * DMODEL;
#pragma unroll
    for (int i = 0; i < 4; ++i) {
        int c = threadIdx.x + i * 256;
        po[c] = (hv[i] - mu) * rstd * w[c] + b[c];
    }
}

// ---------------------------------------------------------------------------
// Launch
// ---------------------------------------------------------------------------
extern "C" void kernel_launch(void* const* d_in, const int* in_sizes, int n_in,
                              void* d_out, int out_size, void* d_ws, size_t ws_size,
                              hipStream_t stream) {
    const float* x      = (const float*)d_in[0];   // (8, 4096, 1024)
    const float* Win    = (const float*)d_in[1];   // (1024, 2048)
    const float* bin    = (const float*)d_in[2];   // (2048)
    const float* conv_w = (const float*)d_in[3];   // (1024, 1, 3)
    const float* conv_b = (const float*)d_in[4];   // (1024)
    const float* lga    = (const float*)d_in[5];   // (1024)
    const float* ssm_b  = (const float*)d_in[6];
    const float* ssm_c  = (const float*)d_in[7];
    const float* ssm_d  = (const float*)d_in[8];
    const float* Wout   = (const float*)d_in[9];   // (1024, 1024)
    const float* bout   = (const float*)d_in[10];  // (1024)
    const float* ln_w   = (const float*)d_in[11];
    const float* ln_b   = (const float*)d_in[12];
    float* out = (float*)d_out;

    // Workspace layout (256B aligned):
    char* ws = (char*)d_ws;
    size_t off = 0;
    auto take = [&](size_t bytes) {
        char* p = ws + off;
        off = (off + bytes + 255) & ~(size_t)255;
        return p;
    };
    __bf16* xb     = (__bf16*)take((size_t)NTOK * DMODEL * sizeof(__bf16));  // 64 MB
    __bf16* Wt_in  = (__bf16*)take((size_t)2048 * 1024 * sizeof(__bf16));    // 4 MB
    __bf16* Wt_out = (__bf16*)take((size_t)1024 * 1024 * sizeof(__bf16));    // 2 MB
    float*  u_buf  = (float*) take((size_t)NTOK * DMODEL * sizeof(float));   // 128 MB
    float*  v_buf  = (float*) take((size_t)NTOK * DMODEL * sizeof(float));   // 128 MB (reused as GEMM2 out)
    __bf16* z_buf  = (__bf16*)take((size_t)NTOK * DMODEL * sizeof(__bf16));  // 64 MB
    (void)ws_size;

    // 1) convert activations + weights to bf16 (weights transposed to [N][K])
    {
        int n4 = NTOK * DMODEL / 4;
        cvt_x_kernel<<<(n4 + 255) / 256, 256, 0, stream>>>(x, xb, n4);
        int n1 = 1024 * 2048;
        cvt_transpose_kernel<<<(n1 + 255) / 256, 256, 0, stream>>>(Win, Wt_in, 1024, 2048);
        int n2 = 1024 * 1024;
        cvt_transpose_kernel<<<(n2 + 255) / 256, 256, 0, stream>>>(Wout, Wt_out, 1024, 1024);
    }

    // 2) in-proj GEMM: xb @ Wt_in -> u, silu(v)
    {
        dim3 grid(NTOK / 128, 2048 / 128);
        gemm_wmma_kernel<0><<<grid, 256, 0, stream>>>(
            xb, Wt_in, bin, u_buf, v_buf, NTOK, 2048, 1024);
    }

    // 3) fused conv + scan + gate -> z (bf16)
    {
        int nch = BATCH * DMODEL;
        conv_scan_gate_kernel<<<(nch + 255) / 256, 256, 0, stream>>>(
            v_buf, u_buf, conv_w, conv_b, lga, ssm_b, ssm_c, ssm_d, z_buf);
    }

    // 4) out-proj GEMM: z @ Wt_out + bout -> v_buf (fp32 pre-LN)
    {
        dim3 grid(NTOK / 128, 1024 / 128);
        gemm_wmma_kernel<1><<<grid, 256, 0, stream>>>(
            z_buf, Wt_out, bout, v_buf, nullptr, NTOK, 1024, 1024);
    }

    // 5) residual + LayerNorm -> d_out
    ln_kernel<<<NTOK, 256, 0, stream>>>(v_buf, x, ln_w, ln_b, out);
}
